// BailingMoE_87333864996961
// MI455X (gfx1250) — compile-verified, matrix operand order checked by reading, not run on hardware
//
#include <hip/hip_runtime.h>
#include <hip/hip_bf16.h>

// ---------------- problem constants ----------------
#define T_TOK 8192   // tokens
#define HDIM  2048   // hidden
#define NEXP  32     // experts
#define TOPK  4
#define FDIM  1408   // routed expert intermediate (also each shared half)
#define FSDIM 2816   // shared intermediate (2 * FDIM)
#define ROUTED_SCALE 1.0f

// ---------------- WMMA types ----------------
typedef __attribute__((ext_vector_type(16))) __bf16 v16bf;
typedef __attribute__((ext_vector_type(8)))  float  v8f;

union Frag {
    uint4 q[2];   // two 16B chunks = 16 bf16 per lane
    v16bf v;
};

// literal-arg macro so the builtin's bool/short modifiers stay ICEs
#define WMMA_BF16(A, B, C) \
    __builtin_amdgcn_wmma_f32_16x16x32_bf16(false, (A).v, false, (B).v, (short)0, (C), false, false)

// hardware f32->bf16 (v_cvt_bf16_f32 on gfx1250; clang soft-falls-back otherwise)
__device__ __forceinline__ unsigned short f2bf(float f) {
    return __builtin_bit_cast(unsigned short, (__bf16)f);
}
// packed pair: lo -> bits[15:0], hi -> bits[31:16].  Inline asm keeps the exact
// (row k, row k+1) pairing opaque so the optimizer can't re-SLP it into
// consecutive-component converts + half-swizzles.
__device__ __forceinline__ unsigned cvt2bf(float lo, float hi) {
#if defined(__gfx1250__)
    unsigned r;
    asm("v_cvt_pk_bf16_f32 %0, %1, %2" : "=v"(r) : "v"(lo), "v"(hi));
    return r;
#elif __has_builtin(__builtin_amdgcn_cvt_pk_bf16_f32)
    auto r = __builtin_amdgcn_cvt_pk_bf16_f32(lo, hi);
    return __builtin_bit_cast(unsigned, r);
#else
    return ((unsigned)f2bf(hi) << 16) | (unsigned)f2bf(lo);
#endif
}
__device__ __forceinline__ uint2 pack4bf(float4 v) {
    uint2 o;
    o.x = cvt2bf(v.x, v.y);
    o.y = cvt2bf(v.z, v.w);
    return o;
}

// ---------------- LDS tiling ----------------
constexpr int MT = 32;                 // token tile (2 WMMA M-tiles)
constexpr int XS = HDIM + 8;           // 2056 halfs, rows 16B-aligned
constexpr int HS = FDIM + 8;           // 1416 halfs, rows 16B-aligned
constexpr int BS = 40;                 // B-tile row stride (80B, 16B-aligned)
constexpr int BTILE = 128 * BS;        // one B buffer (halfs)
constexpr size_t SMEM_BYTES =
    ((size_t)MT * XS + (size_t)MT * HS + (size_t)4 * BTILE) * sizeof(unsigned short);
static_assert(SMEM_BYTES < 320u * 1024u, "LDS budget (320KB/WGP on CDNA5)");

// =====================================================================
// Router: 1 wave32 per token, lane == expert.  softmax -> top4 -> renorm,
// scatter straight into per-expert token lists.
// =====================================================================
__global__ __launch_bounds__(256)
void router_kernel(const float* __restrict__ x, const float* __restrict__ gw,
                   int* __restrict__ counts, int* __restrict__ expIds,
                   float* __restrict__ expW)
{
    const int wave = threadIdx.x >> 5, lane = threadIdx.x & 31;
    const int t = blockIdx.x * 8 + wave;

    const float4* xr = (const float4*)(x + (size_t)t * HDIM);
    const float4* gr = (const float4*)(gw + (size_t)lane * HDIM);
    float acc = 0.f;
#pragma unroll 4
    for (int i = 0; i < HDIM / 4; ++i) {
        float4 a = xr[i], b = gr[i];
        acc += a.x * b.x + a.y * b.y + a.z * b.z + a.w * b.w;
    }

    // wave32 softmax across experts
    float m = acc;
    for (int off = 16; off; off >>= 1) m = fmaxf(m, __shfl_xor(m, off, 32));
    float p = __expf(acc - m);
    float s = p;
    for (int off = 16; off; off >>= 1) s += __shfl_xor(s, off, 32);
    p /= s;

    // iterative top-4 (exclude winners); ties -> lowest index like jax top_k
    float pw = p;
    float v0 = 0, v1 = 0, v2 = 0, v3 = 0;
    int   i0 = 0, i1 = 0, i2 = 0, i3 = 0;
#pragma unroll
    for (int k = 0; k < TOPK; ++k) {
        float bv = pw; int bi = lane;
        for (int off = 16; off; off >>= 1) {
            float ov = __shfl_xor(bv, off, 32);
            int   oi = __shfl_xor(bi, off, 32);
            if (ov > bv || (ov == bv && oi < bi)) { bv = ov; bi = oi; }
        }
        if      (k == 0) { v0 = bv; i0 = bi; }
        else if (k == 1) { v1 = bv; i1 = bi; }
        else if (k == 2) { v2 = bv; i2 = bi; }
        else             { v3 = bv; i3 = bi; }
        if (lane == bi) pw = -1.f;
    }
    const float s4 = v0 + v1 + v2 + v3;
    if (lane < TOPK) {
        float val = (lane == 0 ? v0 : lane == 1 ? v1 : lane == 2 ? v2 : v3) / s4;
        int   e   = (lane == 0 ? i0 : lane == 1 ? i1 : lane == 2 ? i2 : i3);
        int pos = atomicAdd(&counts[e], 1);
        expIds[(size_t)e * T_TOK + pos] = t;
        expW [(size_t)e * T_TOK + pos] = val * ROUTED_SCALE;
    }
}

// =====================================================================
// Fused expert MLP: gate_up GEMM + SiLU*mul (h parked in LDS) + down GEMM,
// weighted atomic accumulate into out.  Double-buffered weight tiles; each
// thread owns a 2(K) x 8(N) slice so bf16 pairs pack into single b32 LDS
// stores via v_cvt_pk (k, k+1 contiguous in the [n][k] tile).
// =====================================================================
__global__ __launch_bounds__(256, 1)
void moe_mlp_kernel(const float* __restrict__ x,
                    const float* __restrict__ wguBase,
                    const float* __restrict__ wdBase,
                    const int*   __restrict__ expIds,
                    const float* __restrict__ expW,
                    const int*   __restrict__ counts,
                    float*       __restrict__ out,
                    int ldgu, int gOff, int uOff,
                    long long wguStride, long long wdStride, int fixedCnt)
{
    extern __shared__ unsigned short smem[];
    unsigned short* Xs = smem;                   // [MT][XS]  bf16 activations
    unsigned short* Hs = Xs + MT * XS;           // [MT][HS]  bf16 silu(g)*u
    unsigned short* Bbuf = Hs + MT * HS;         // 4 x [128][BS] ping-pong tiles
    __shared__ int   sTok[MT];
    __shared__ float sW[MT];

    const int tid  = threadIdx.x;
    const int wave = tid >> 5, lane = tid & 31;
    const int e    = blockIdx.x >> 8;            // routed: expert; shared: 0
    const int tile = blockIdx.x & 255;
    const int m0   = tile * MT;

    const float* wgu = wguBase + (size_t)e * (size_t)wguStride;
    const float* wd  = wdBase  + (size_t)e * (size_t)wdStride;

    const int cnt = expIds ? counts[e] : fixedCnt;
    if (m0 >= cnt) return;

    if (tid < MT) {
        int gm = m0 + tid;
        int tok; float w;
        if (expIds) {
            const int*   ids = expIds + (size_t)e * T_TOK;
            const float* wts = expW  + (size_t)e * T_TOK;
            if (gm < cnt) { tok = ids[gm]; w = wts[gm]; }
            else          { tok = ids[m0]; w = 0.f; }     // pad rows contribute 0
        } else {
            tok = (gm < cnt) ? gm : (cnt - 1);
            w   = (gm < cnt) ? 1.f : 0.f;
        }
        sTok[tid] = tok; sW[tid] = w;
    }
    __syncthreads();

    // gather X tile -> bf16 LDS (float4 / b128 loads, packed b64 stores)
    for (int i = tid; i < MT * (HDIM / 4); i += 256) {
        int r  = i >> 9;                 // / (HDIM/4)
        int c4 = i & (HDIM / 4 - 1);
        float4 v = ((const float4*)(x + (size_t)sTok[r] * HDIM))[c4];
        *(uint2*)&Xs[r * XS + c4 * 4] = pack4bf(v);
    }

    const int r0   = lane & 15;
    const int kh   = (lane >> 4) << 3;          // K-half select: 0 or 8
    const int ncol = (wave << 4) + (lane & 15); // this lane's local output col
    const int hi   = (lane >> 4) << 3;          // C row offset: 0 or 8

    // per-thread slice of a 128(n) x 32(k) tile: rows 2*tkp,2*tkp+1 x cols tn8..+7
    const int tn8 = (tid & 15) << 3;            // n base (0..120, step 8)
    const int tkp = tid >> 4;                   // k-pair index (0..15)

    // packed store of one column-pair slice: B[(n)(..+3)][2tkp..2tkp+1]
    auto stPair = [&](unsigned short* B, int nb4, float4 lo, float4 hi) {
        *(unsigned*)&B[(tn8 + nb4 + 0) * BS + 2 * tkp] = cvt2bf(lo.x, hi.x);
        *(unsigned*)&B[(tn8 + nb4 + 1) * BS + 2 * tkp] = cvt2bf(lo.y, hi.y);
        *(unsigned*)&B[(tn8 + nb4 + 2) * BS + 2 * tkp] = cvt2bf(lo.z, hi.z);
        *(unsigned*)&B[(tn8 + nb4 + 3) * BS + 2 * tkp] = cvt2bf(lo.w, hi.w);
    };

    // ---------------- Stage 1: gu = X @ Wgu ; h = silu(g)*u -> Hs -------------
    for (int nb = 0; nb < FDIM; nb += 128) {
        float4 g0a, g0b, g1a, g1b, u0a, u0b, u1a, u1b;

        auto loadGU = [&](int k0) {
            const float* p0 = wgu + (size_t)(k0 + 2 * tkp) * (size_t)ldgu
                                  + (size_t)(nb + tn8);
            const float* p1 = p0 + ldgu;
            g0a = *(const float4*)(p0 + gOff);  g0b = *(const float4*)(p0 + gOff + 4);
            g1a = *(const float4*)(p1 + gOff);  g1b = *(const float4*)(p1 + gOff + 4);
            u0a = *(const float4*)(p0 + uOff);  u0b = *(const float4*)(p0 + uOff + 4);
            u1a = *(const float4*)(p1 + uOff);  u1b = *(const float4*)(p1 + uOff + 4);
        };
        auto storeGU = [&](unsigned short* Bg, unsigned short* Bu) {
            stPair(Bg, 0, g0a, g1a);  stPair(Bg, 4, g0b, g1b);
            stPair(Bu, 0, u0a, u1a);  stPair(Bu, 4, u0b, u1b);
        };

        v8f aG0 = {}; v8f aG1 = {}; v8f aU0 = {}; v8f aU1 = {};
        loadGU(0);
        storeGU(Bbuf, Bbuf + BTILE);            // prime buffer 0
        int cur = 0;
        for (int k0 = 0; k0 < HDIM; k0 += 32) {
            __syncthreads();                    // buf[cur] ready; buf[cur^1] free
            const bool more = (k0 + 32 < HDIM);
            if (more) loadGU(k0 + 32);          // global loads fly under WMMA

            unsigned short* Bg = Bbuf + cur * 2 * BTILE;
            unsigned short* Bu = Bg + BTILE;
            Frag a0, a1, bg, bu;
            const unsigned short* pa0 = &Xs[r0 * XS + k0 + kh];
            a0.q[0] = *(const uint4*)pa0;       a0.q[1] = *(const uint4*)(pa0 + 16);
            const unsigned short* pa1 = &Xs[(r0 + 16) * XS + k0 + kh];
            a1.q[0] = *(const uint4*)pa1;       a1.q[1] = *(const uint4*)(pa1 + 16);
            const unsigned short* pbg = &Bg[ncol * BS + kh];
            bg.q[0] = *(const uint4*)pbg;       bg.q[1] = *(const uint4*)(pbg + 16);
            const unsigned short* pbu = &Bu[ncol * BS + kh];
            bu.q[0] = *(const uint4*)pbu;       bu.q[1] = *(const uint4*)(pbu + 16);

            aG0 = WMMA_BF16(a0, bg, aG0);
            aG1 = WMMA_BF16(a1, bg, aG1);
            aU0 = WMMA_BF16(a0, bu, aU0);
            aU1 = WMMA_BF16(a1, bu, aU1);

            if (more) {
                unsigned short* nBg = Bbuf + (cur ^ 1) * 2 * BTILE;
                storeGU(nBg, nBg + BTILE);
            }
            cur ^= 1;
        }
        const int col = nb + ncol;
#pragma unroll
        for (int v = 0; v < 8; ++v) {
            float g = aG0[v], u = aU0[v];
            Hs[(v + hi) * HS + col]      = f2bf(u * (g / (1.f + __expf(-g))));
            g = aG1[v]; u = aU1[v];
            Hs[(16 + v + hi) * HS + col] = f2bf(u * (g / (1.f + __expf(-g))));
        }
    }

    // ---------------- Stage 2: out += w * (h @ Wd) ----------------------------
    for (int nb = 0; nb < HDIM; nb += 128) {
        float4 d0a, d0b, d1a, d1b;

        auto loadD = [&](int k0) {
            const float* p0 = wd + (size_t)(k0 + 2 * tkp) * HDIM + (size_t)(nb + tn8);
            const float* p1 = p0 + HDIM;
            d0a = *(const float4*)p0;  d0b = *(const float4*)(p0 + 4);
            d1a = *(const float4*)p1;  d1b = *(const float4*)(p1 + 4);
        };
        auto storeD = [&](unsigned short* Bd) {
            stPair(Bd, 0, d0a, d1a);  stPair(Bd, 4, d0b, d1b);
        };

        v8f aO0 = {}; v8f aO1 = {};
        loadD(0);
        storeD(Bbuf);                           // buffer 0
        int cur = 0;
        for (int k0 = 0; k0 < FDIM; k0 += 32) {
            __syncthreads();   // 1st iter also makes Hs writes visible
            const bool more = (k0 + 32 < FDIM);
            if (more) loadD(k0 + 32);

            unsigned short* Bd = Bbuf + cur * 2 * BTILE;
            Frag a0, a1, bd;
            const unsigned short* pa0 = &Hs[r0 * HS + k0 + kh];
            a0.q[0] = *(const uint4*)pa0;       a0.q[1] = *(const uint4*)(pa0 + 16);
            const unsigned short* pa1 = &Hs[(r0 + 16) * HS + k0 + kh];
            a1.q[0] = *(const uint4*)pa1;       a1.q[1] = *(const uint4*)(pa1 + 16);
            const unsigned short* pbd = &Bd[ncol * BS + kh];
            bd.q[0] = *(const uint4*)pbd;       bd.q[1] = *(const uint4*)(pbd + 16);

            aO0 = WMMA_BF16(a0, bd, aO0);
            aO1 = WMMA_BF16(a1, bd, aO1);

            if (more) storeD(Bbuf + (cur ^ 1) * 2 * BTILE);
            cur ^= 1;
        }
        const int col = nb + ncol;
#pragma unroll
        for (int v = 0; v < 8; ++v) {
            int m = v + hi;
            unsafeAtomicAdd(&out[(size_t)sTok[m] * HDIM + col], sW[m] * aO0[v]);
            m += 16;
            unsafeAtomicAdd(&out[(size_t)sTok[m] * HDIM + col], sW[m] * aO1[v]);
        }
        __syncthreads();   // all reads of Hs/Bbuf done before next nb re-primes
    }
}

// =====================================================================
extern "C" void kernel_launch(void* const* d_in, const int* in_sizes, int n_in,
                              void* d_out, int out_size, void* d_ws, size_t ws_size,
                              hipStream_t stream) {
    const float* x        = (const float*)d_in[0];  // [T,H]
    const float* gate_w   = (const float*)d_in[1];  // [E,H]
    const float* w_gu     = (const float*)d_in[2];  // [E,H,2F]
    const float* w_dn     = (const float*)d_in[3];  // [E,F,H]
    const float* s_gu     = (const float*)d_in[4];  // [H,2FS]
    const float* s_dn     = (const float*)d_in[5];  // [FS,H]
    float*       out      = (float*)d_out;          // [T,H]

    // workspace: counts | expert token ids | expert token weights
    char*  ws     = (char*)d_ws;
    int*   counts = (int*)ws;                              // E ints
    int*   expIds = (int*)(ws + 256);                      // E*T ints
    float* expW   = (float*)(ws + 256 + (size_t)NEXP * T_TOK * 4);

    hipMemsetAsync(counts, 0, NEXP * sizeof(int), stream);
    hipMemsetAsync(out, 0, (size_t)out_size * sizeof(float), stream);

    // 1) router + scatter into per-expert lists
    router_kernel<<<T_TOK / 8, 256, 0, stream>>>(x, gate_w, counts, expIds, expW);

    // 2) routed experts: grid = expert * 256 token-tiles (excess tiles early-exit;
    //    same-expert blocks contiguous for L2 weight reuse)
    moe_mlp_kernel<<<NEXP * 256, 256, SMEM_BYTES, stream>>>(
        x, w_gu, w_dn, expIds, expW, counts, out,
        /*ldgu=*/2 * FDIM, /*gOff=*/0, /*uOff=*/FDIM,
        /*wguStride=*/(long long)HDIM * 2 * FDIM, /*wdStride=*/(long long)FDIM * HDIM,
        /*fixedCnt=*/0);

    // 3) shared expert as two F=1408 pseudo-experts over all tokens, weight 1
    moe_mlp_kernel<<<256, 256, SMEM_BYTES, stream>>>(
        x, s_gu, s_dn, nullptr, nullptr, nullptr, out,
        /*ldgu=*/2 * FSDIM, /*gOff=*/0, /*uOff=*/FSDIM,
        0, 0, /*fixedCnt=*/T_TOK);
    moe_mlp_kernel<<<256, 256, SMEM_BYTES, stream>>>(
        x, s_gu, s_dn + (size_t)FDIM * HDIM, nullptr, nullptr, nullptr, out,
        /*ldgu=*/2 * FSDIM, /*gOff=*/FDIM, /*uOff=*/FSDIM + FDIM,
        0, 0, /*fixedCnt=*/T_TOK);
}
